// MultiLayerMamba_39238821216796
// MI455X (gfx1250) — compile-verified, matrix operand order checked by reading, not run on hardware
//
#include <hip/hip_runtime.h>
#include <hip/hip_bf16.h>

#define B_SZ     8
#define SEQ      2048
#define D_MODEL  1024
#define N_LAYER  2
#define D_INNER  2048
#define D_STATE  16
#define D_CONV   4
#define DT_RANK  64
#define MROWS    (B_SZ*SEQ)   // 16384 token rows

typedef __attribute__((ext_vector_type(16))) __bf16        v16bf;
typedef __attribute__((ext_vector_type(8)))  float         v8f;
typedef __attribute__((ext_vector_type(4)))  unsigned int  u32x4;

static __device__ __forceinline__ unsigned short f2bf(float f) {
    unsigned int u = __float_as_uint(f);
    u += 0x7FFFu + ((u >> 16) & 1u);          // round-to-nearest-even
    return (unsigned short)(u >> 16);
}
static __device__ __forceinline__ float sigm(float x) { return 1.f / (1.f + __expf(-x)); }

// ---------------- fp32 -> bf16 conversion (weights / small tensors) ----------------
__global__ void cvt_f32_bf16(const float* __restrict__ src, unsigned short* __restrict__ dst, int n) {
    int i = blockIdx.x * 256 + threadIdx.x;
    if (i < n) dst[i] = f2bf(src[i]);
}

// ---------------- residual add + layernorm -> bf16 activations ----------------
__global__ __launch_bounds__(256) void add_ln_kernel(
    const float* __restrict__ x, float* __restrict__ res,
    const float* __restrict__ w, const float* __restrict__ b,
    unsigned short* __restrict__ h, int has_res)
{
    __shared__ float s1[256], s2[256];
    int m = blockIdx.x, t = threadIdx.x;
    const float* xr = x + (size_t)m * D_MODEL;
    float* rr = res + (size_t)m * D_MODEL;
    float v[4], sum = 0.f, sq = 0.f;
#pragma unroll
    for (int j = 0; j < 4; j++) {
        float xv = xr[t + 256 * j];
        if (has_res) xv += rr[t + 256 * j];
        v[j] = xv; sum += xv; sq += xv * xv;
    }
    s1[t] = sum; s2[t] = sq; __syncthreads();
    for (int off = 128; off > 0; off >>= 1) {
        if (t < off) { s1[t] += s1[t + off]; s2[t] += s2[t + off]; }
        __syncthreads();
    }
    float mu   = s1[0] * (1.f / D_MODEL);
    float var  = s2[0] * (1.f / D_MODEL) - mu * mu;
    float rstd = rsqrtf(var + 1e-5f);
    unsigned short* hr = h + (size_t)m * D_MODEL;
#pragma unroll
    for (int j = 0; j < 4; j++) {
        int c = t + 256 * j;
        rr[c] = v[j];
        hr[c] = f2bf((v[j] - mu) * rstd * w[c] + b[c]);
    }
}

// ---------------- WMMA bf16 GEMM: C[M,N] = A[M,K] * W[N,K]^T ----------------
// Double-buffered LDS, staged via GLOBAL_LOAD_ASYNC_TO_LDS_B128 (ASYNCcnt).
// EPI: 0 = plain store, 1 = softplus(acc + bias[n])
#define BM 128
#define BN 64
#define BK 32
#define LDT 48   // LDS row stride in ushorts (96B: 16B aligned, conflict-reducing)

template<int EPI>
__global__ __launch_bounds__(256) void gemm_bf16(
    const unsigned short* __restrict__ A,   // M x K (bf16)
    const unsigned short* __restrict__ Bw,  // N x K (bf16, weight rows = B^T tiles)
    const float* __restrict__ bias,
    float* __restrict__ C, int M, int N, int K, int ldc)
{
    __shared__ __align__(16) unsigned short As[2][BM * LDT];
    __shared__ __align__(16) unsigned short Bs[2][BN * LDT];

    int t = threadIdx.x;
    int lane = t & 31, wave = t >> 5;
    int wm = (wave & 3) * 32;        // wave M-offset inside block tile (4 waves)
    int wn = (wave >> 2) * 32;       // wave N-offset inside block tile (2 waves)
    int m0 = blockIdx.y * BM;
    int n0 = blockIdx.x * BN;
    int kb = (lane & 16) ? 8 : 0;    // K phase for lanes 16..31 per ISA layout
    int rA = lane & 15;              // row (A) / column (B) index within 16

    // per-thread async-staging chunks: A rows t/4 and t/4+64, B row t/4 (8 ushorts each)
    int arow0 = t >> 2, arow1 = (t >> 2) + 64;
    int col   = (t & 3) * 8;
    int brow  = n0 + (t >> 2);
    if (brow > N - 1) brow = N - 1;           // clamp (cols >= N never stored)

    const unsigned short* gA0 = A  + (size_t)(m0 + arow0) * K + col;
    const unsigned short* gA1 = A  + (size_t)(m0 + arow1) * K + col;
    const unsigned short* gB  = Bw + (size_t)brow         * K + col;

    unsigned ldsA0 = (unsigned)(size_t)&As[0][arow0 * LDT + col];
    unsigned ldsA1 = (unsigned)(size_t)&As[0][arow1 * LDT + col];
    unsigned ldsB  = (unsigned)(size_t)&Bs[0][(t >> 2) * LDT + col];
    const unsigned aStride = BM * LDT * 2;    // bytes between A buffers
    const unsigned bStride = BN * LDT * 2;    // bytes between B buffers

    auto issue_tile = [&](int buf, int k0) {
        unsigned ao = buf ? aStride : 0u;
        unsigned bo = buf ? bStride : 0u;
        asm volatile("global_load_async_to_lds_b128 %0, %1, off"
            :: "v"(ldsA0 + ao), "v"((unsigned long long)(size_t)(gA0 + k0)) : "memory");
        asm volatile("global_load_async_to_lds_b128 %0, %1, off"
            :: "v"(ldsA1 + ao), "v"((unsigned long long)(size_t)(gA1 + k0)) : "memory");
        asm volatile("global_load_async_to_lds_b128 %0, %1, off"
            :: "v"(ldsB + bo),  "v"((unsigned long long)(size_t)(gB + k0))  : "memory");
    };

    v8f acc[2][2] = {};

    issue_tile(0, 0);                          // prologue: tile 0 in flight
    int cur = 0;
    for (int k0 = 0; k0 < K; k0 += BK) {
        bool hasnext = (k0 + BK) < K;
        if (hasnext) {
            issue_tile(cur ^ 1, k0 + BK);      // prefetch next tile into other buffer
            // in-order completion: wait for previous 3, leave the 3 just issued in flight
            asm volatile("s_wait_asynccnt 0x3" ::: "memory");
        } else {
            asm volatile("s_wait_asynccnt 0x0" ::: "memory");
        }
        __syncthreads();                       // tile `cur` visible to all waves

        union Frag { v16bf v; u32x4 q[2]; };
        Frag a[2], bb[2];
#pragma unroll
        for (int i = 0; i < 2; i++) {
            const unsigned short* pa = &As[cur][(wm + i * 16 + rA) * LDT];
            a[i].q[0] = *(const u32x4*)(pa + kb);          // K in [kb, kb+8)
            a[i].q[1] = *(const u32x4*)(pa + 16 + kb);     // K in [16+kb, 16+kb+8)
            const unsigned short* pb = &Bs[cur][(wn + i * 16 + rA) * LDT];
            bb[i].q[0] = *(const u32x4*)(pb + kb);
            bb[i].q[1] = *(const u32x4*)(pb + 16 + kb);
        }
#pragma unroll
        for (int i = 0; i < 2; i++)
#pragma unroll
            for (int j = 0; j < 2; j++)
                acc[i][j] = __builtin_amdgcn_wmma_f32_16x16x32_bf16(
                    false, a[i].v, false, bb[j].v, (short)0, acc[i][j], false, false);
        __syncthreads();                       // done reading `cur` before it is re-filled
        cur ^= 1;
    }

    // store: C/D layout — VGPR r holds M = r + 8*(lane>=16), N = lane&15
    int hi = (lane & 16) ? 8 : 0;
#pragma unroll
    for (int i = 0; i < 2; i++)
#pragma unroll
        for (int j = 0; j < 2; j++) {
            int colg = n0 + wn + j * 16 + rA;
            if (colg >= N) continue;
            float bv = 0.f;
            if constexpr (EPI == 1) bv = bias[colg];
#pragma unroll
            for (int r = 0; r < 8; r++) {
                int rowg = m0 + wm + i * 16 + hi + r;
                float v = acc[i][j][r];
                if constexpr (EPI == 1) {
                    v += bv;
                    v = (v > 20.f) ? v : log1pf(__expf(v));   // softplus
                }
                C[(size_t)rowg * ldc + colg] = v;
            }
        }
}

// ---------------- causal depthwise conv (D_CONV=4) + SiLU ----------------
__global__ void conv_silu_kernel(const float* __restrict__ xz,
                                 const float* __restrict__ cw, const float* __restrict__ cb,
                                 float* __restrict__ xc, unsigned short* __restrict__ xcb)
{
    size_t i = (size_t)blockIdx.x * 256 + threadIdx.x;
    if (i >= (size_t)MROWS * D_INNER) return;
    int d = (int)(i % D_INNER);
    size_t ml = i / D_INNER;          // b*SEQ + l
    int l = (int)(ml % SEQ);
    size_t rowb = ml - l;             // b*SEQ
    float y = cb[d];
#pragma unroll
    for (int k = 0; k < D_CONV; k++) {
        int ll = l - (D_CONV - 1) + k;
        if (ll >= 0)
            y += xz[(rowb + ll) * (size_t)(2 * D_INNER) + d] * cw[d * D_CONV + k];
    }
    float s = y * sigm(y);
    xc[i] = s;
    xcb[i] = f2bf(s);
}

// ---------------- extract dt_low (first 64 cols of xdbl) as bf16 ----------------
__global__ void dtlow_kernel(const float* __restrict__ xdbl, unsigned short* __restrict__ dl) {
    int i = blockIdx.x * 256 + threadIdx.x;
    if (i >= MROWS * DT_RANK) return;
    int m = i / DT_RANK, r = i % DT_RANK;
    dl[i] = f2bf(xdbl[(size_t)m * 96 + r]);
}

// ---------------- selective scan: 1 thread per (b, d_inner) channel ----------------
#define SCHUNK 64
__global__ __launch_bounds__(256) void scan_kernel(
    const float* __restrict__ u,      // xc   (MROWS, D_INNER)
    const float* __restrict__ dt,     // dt   (MROWS, D_INNER)
    const float* __restrict__ xdbl,   // (MROWS, 96): B at col 64, C at col 80
    const float* __restrict__ A_log,  // (D_INNER, 16)
    const float* __restrict__ Dp,     // (D_INNER)
    float* __restrict__ y, int ystride)
{
    __shared__ float Bsh[SCHUNK][D_STATE];
    __shared__ float Csh[SCHUNK][D_STATE];
    int t = threadIdx.x;
    int b = blockIdx.x / (D_INNER / 256);
    int d = (blockIdx.x % (D_INNER / 256)) * 256 + t;

    float Av[D_STATE];
#pragma unroll
    for (int s = 0; s < D_STATE; s++) Av[s] = -__expf(A_log[d * D_STATE + s]);
    float dpar = Dp[d];
    float h[D_STATE];
#pragma unroll
    for (int s = 0; s < D_STATE; s++) h[s] = 0.f;

    size_t rowbase = (size_t)b * SEQ;
    for (int l0 = 0; l0 < SEQ; l0 += SCHUNK) {
        __syncthreads();
        for (int j = t; j < SCHUNK * 32; j += 256) {
            int i = j >> 5, c = j & 31;
            float v = xdbl[(rowbase + l0 + i) * 96 + 64 + c];
            if (c < D_STATE) Bsh[i][c] = v; else Csh[i][c - D_STATE] = v;
        }
        __syncthreads();
        for (int i = 0; i < SCHUNK; i++) {
            size_t idx = (rowbase + l0 + i) * (size_t)D_INNER + d;
            float ut = u[idx], dtt = dt[idx];
            float dtu = dtt * ut;
            float acc = 0.f;
#pragma unroll
            for (int s = 0; s < D_STATE; s++) {
                float dA = __expf(dtt * Av[s]);
                h[s] = dA * h[s] + dtu * Bsh[i][s];
                acc += h[s] * Csh[i][s];
            }
            y[(rowbase + l0 + i) * (size_t)ystride + d] = acc + ut * dpar;
        }
    }
}

// ---------------- y * silu(z) gate -> bf16 for out_proj ----------------
__global__ void gate_kernel(const float* __restrict__ xz, unsigned short* __restrict__ yb) {
    size_t i = (size_t)blockIdx.x * 256 + threadIdx.x;
    if (i >= (size_t)MROWS * D_INNER) return;
    size_t m = i / D_INNER; int d = (int)(i % D_INNER);
    float yv = xz[m * (size_t)(2 * D_INNER) + d];            // scan output (reused xi slot)
    float z  = xz[m * (size_t)(2 * D_INNER) + D_INNER + d];
    yb[i] = f2bf(yv * z * sigm(z));
}

// =======================================================================
extern "C" void kernel_launch(void* const* d_in, const int* in_sizes, int n_in,
                              void* d_out, int out_size, void* d_ws, size_t ws_size,
                              hipStream_t stream)
{
    const float* x_in   = (const float*)d_in[0];
    const float* in_w   = (const float*)d_in[1];   // (2, 4096, 1024)
    const float* conv_w = (const float*)d_in[2];   // (2, 2048, 4)
    const float* conv_b = (const float*)d_in[3];   // (2, 2048)
    const float* xp_w   = (const float*)d_in[4];   // (2, 96, 2048)
    const float* dt_w   = (const float*)d_in[5];   // (2, 2048, 64)
    const float* dt_b   = (const float*)d_in[6];   // (2, 2048)
    const float* A_log  = (const float*)d_in[7];   // (2, 2048, 16)
    const float* D_par  = (const float*)d_in[8];   // (2, 2048)
    const float* out_w  = (const float*)d_in[9];   // (2, 1024, 2048)
    const float* nw     = (const float*)d_in[10];  // (2, 1024)
    const float* nb     = (const float*)d_in[11];  // (2, 1024)

    char* p = (char*)d_ws;
    auto carve = [&](size_t bytes) -> void* {
        void* r = (void*)p; p += (bytes + 255) & ~(size_t)255; return r;
    };
    const size_t M = MROWS;
    float*          res    = (float*)carve(M * D_MODEL * 4);
    float*          xz     = (float*)carve(M * 2 * D_INNER * 4);   // xi|z, xi slot reused for y
    float*          xc     = (float*)carve(M * D_INNER * 4);
    float*          dtbuf  = (float*)carve(M * D_INNER * 4);
    float*          xdbl   = (float*)carve(M * 96 * 4);
    float*          xnext  = (float*)carve(M * D_MODEL * 4);
    unsigned short* h_bf   = (unsigned short*)carve(M * D_MODEL * 2);
    unsigned short* xc_bf  = (unsigned short*)carve(M * D_INNER * 2);
    unsigned short* y_bf   = (unsigned short*)carve(M * D_INNER * 2);
    unsigned short* dtl_bf = (unsigned short*)carve(M * DT_RANK * 2);
    unsigned short* inw_bf = (unsigned short*)carve((size_t)(2 * D_INNER) * D_MODEL * 2);
    unsigned short* xpw_bf = (unsigned short*)carve((size_t)96 * D_INNER * 2);
    unsigned short* dtw_bf = (unsigned short*)carve((size_t)D_INNER * DT_RANK * 2);
    unsigned short* ow_bf  = (unsigned short*)carve((size_t)D_MODEL * D_INNER * 2);

    const float* xcur = x_in;
    for (int l = 0; l < N_LAYER; l++) {
        // weight conversions (per-layer slices)
        int n1 = 2 * D_INNER * D_MODEL;
        cvt_f32_bf16<<<(n1 + 255) / 256, 256, 0, stream>>>(in_w + (size_t)l * n1, inw_bf, n1);
        int n2 = 96 * D_INNER;
        cvt_f32_bf16<<<(n2 + 255) / 256, 256, 0, stream>>>(xp_w + (size_t)l * n2, xpw_bf, n2);
        int n3 = D_INNER * DT_RANK;
        cvt_f32_bf16<<<(n3 + 255) / 256, 256, 0, stream>>>(dt_w + (size_t)l * n3, dtw_bf, n3);
        int n4 = D_MODEL * D_INNER;
        cvt_f32_bf16<<<(n4 + 255) / 256, 256, 0, stream>>>(out_w + (size_t)l * n4, ow_bf, n4);

        // residual + layernorm
        add_ln_kernel<<<MROWS, 256, 0, stream>>>(xcur, res, nw + l * D_MODEL, nb + l * D_MODEL,
                                                 h_bf, l > 0 ? 1 : 0);
        // in_proj: (M,1024) x (4096,1024)^T -> xz
        gemm_bf16<0><<<dim3(2 * D_INNER / BN, MROWS / BM), 256, 0, stream>>>(
            h_bf, inw_bf, nullptr, xz, MROWS, 2 * D_INNER, D_MODEL, 2 * D_INNER);
        // causal conv + silu
        conv_silu_kernel<<<(int)((M * D_INNER + 255) / 256), 256, 0, stream>>>(
            xz, conv_w + (size_t)l * D_INNER * D_CONV, conv_b + (size_t)l * D_INNER, xc, xc_bf);
        // x_proj: (M,2048) x (96,2048)^T -> xdbl (N=96, clamped B rows in pad tile)
        gemm_bf16<0><<<dim3((96 + BN - 1) / BN, MROWS / BM), 256, 0, stream>>>(
            xc_bf, xpw_bf, nullptr, xdbl, MROWS, 96, D_INNER, 96);
        // dt_low -> bf16
        dtlow_kernel<<<(MROWS * DT_RANK + 255) / 256, 256, 0, stream>>>(xdbl, dtl_bf);
        // dt_proj + bias + softplus: (M,64) x (2048,64)^T -> dt
        gemm_bf16<1><<<dim3(D_INNER / BN, MROWS / BM), 256, 0, stream>>>(
            dtl_bf, dtw_bf, dt_b + (size_t)l * D_INNER, dtbuf, MROWS, D_INNER, DT_RANK, D_INNER);
        // selective scan (writes y into xi slot of xz, stride 4096)
        scan_kernel<<<B_SZ * (D_INNER / 256), 256, 0, stream>>>(
            xc, dtbuf, xdbl, A_log + (size_t)l * D_INNER * D_STATE,
            D_par + (size_t)l * D_INNER, xz, 2 * D_INNER);
        // gate: y * silu(z) -> bf16
        gate_kernel<<<(int)((M * D_INNER + 255) / 256), 256, 0, stream>>>(xz, y_bf);
        // out_proj: (M,2048) x (1024,2048)^T
        float* outp = (l == N_LAYER - 1) ? (float*)d_out : xnext;
        gemm_bf16<0><<<dim3(D_MODEL / BN, MROWS / BM), 256, 0, stream>>>(
            y_bf, ow_bf, nullptr, outp, MROWS, D_MODEL, D_INNER, D_MODEL);
        xcur = xnext;
    }
}
